// ZonalMoE_72060961292573
// MI455X (gfx1250) — matrix operand order repaired and checked by reading, not compiled, not run on hardware
//
#include <hip/hip_runtime.h>

typedef __bf16 bf16;
typedef __attribute__((ext_vector_type(16))) __bf16 v16bf;
typedef __attribute__((ext_vector_type(8)))  float  v8f;

// ---------------------------------------------------------------------------
// GEMM: C[M,Nn] = act(A[M,K] @ W[K,Nn] + bias)
// Weights pre-transposed+padded: Wt[col][Kp], Kp = ceil32(K), zero padded.
// One wave32 computes a 16x32 output tile = 2 x v_wmma_f32_16x16x32_bf16 per
// 32-wide K step, reusing the A fragment. No branches in the K loop: OOB
// rows/cols are clamped (their D entries are never stored), and K padding is
// absorbed by zeros in Wt.
// Lane layouts (A/B use contiguous K, not the ISA table order):
//   A 16x32 bf16 : lane = M row (l16), K = half*16+e
//   B 32x16 bf16 : lane = N col (l16), K = half*16+e
//   C 16x16 f32  : vgpr r -> M = r + 8*half, N = l16
// ---------------------------------------------------------------------------
__global__ void k_gemm_wmma(const float* __restrict__ A, const bf16* __restrict__ Wt,
                            const float* __restrict__ bias, float* __restrict__ C,
                            int M, int K, int Nn, int act)
{
    int Kp = (K + 31) & ~31;
    int tiles_n = (Nn + 31) >> 5;
    int tiles_m = (M + 15) >> 4;
    int wave = (blockIdx.x * blockDim.x + threadIdx.x) >> 5;   // wave-uniform
    if (wave >= tiles_m * tiles_n) return;
    int row0 = (wave / tiles_n) << 4;
    int col0 = (wave % tiles_n) << 5;
    int lane = threadIdx.x & 31;
    int half = lane >> 4;
    int l16  = lane & 15;

    int arow = row0 + l16;
    int ar   = (arow < M) ? arow : (M - 1);            // clamp: OOB rows never stored
    int bcol0 = col0 + l16;
    int bcol1 = col0 + 16 + l16;
    int bc0 = (bcol0 < Nn) ? bcol0 : (Nn - 1);         // clamp: OOB cols never stored
    int bc1 = (bcol1 < Nn) ? bcol1 : (Nn - 1);

    const float* ap  = A  + (size_t)ar  * K;
    const bf16*  bp0 = Wt + (size_t)bc0 * Kp;
    const bf16*  bp1 = Wt + (size_t)bc1 * Kp;

    v8f acc0 = {}, acc1 = {};
    bool vecA = (K & 3) == 0;                          // 16B-aligned float4 path

    for (int k0 = 0; k0 < Kp; k0 += 32) {
        __builtin_prefetch(ap + k0 + 32);
        v16bf a, b0, b1;
        if (vecA) {
            const float4* pa0 = (const float4*)(ap + k0 + half * 16);
            const float4* pa1 = (const float4*)(ap + k0 + half * 16 + 8);
            float4 x0 = pa0[0], x1 = pa0[1], x2 = pa1[0], x3 = pa1[1];
            a[0] = (bf16)x0.x; a[1] = (bf16)x0.y; a[2]  = (bf16)x0.z; a[3]  = (bf16)x0.w;
            a[4] = (bf16)x1.x; a[5] = (bf16)x1.y; a[6]  = (bf16)x1.z; a[7]  = (bf16)x1.w;
            a[8] = (bf16)x2.x; a[9] = (bf16)x2.y; a[10] = (bf16)x2.z; a[11] = (bf16)x2.w;
            a[12] = (bf16)x3.x; a[13] = (bf16)x3.y; a[14] = (bf16)x3.z; a[15] = (bf16)x3.w;
        } else {
#pragma unroll
            for (int e = 0; e < 16; ++e) {
                int kk = k0 + half * 16 + e;
                a[e] = (bf16)ap[kk];
            }
        }
        b0 = *(const v16bf*)(bp0 + k0 + half * 16);    // 32B contiguous, 32B aligned
        b1 = *(const v16bf*)(bp1 + k0 + half * 16);
        acc0 = __builtin_amdgcn_wmma_f32_16x16x32_bf16(false, a, false, b0,
                                                       (short)0, acc0, false, false);
        acc1 = __builtin_amdgcn_wmma_f32_16x16x32_bf16(false, a, false, b1,
                                                       (short)0, acc1, false, false);
    }

#pragma unroll
    for (int r = 0; r < 8; ++r) {
        int rr = row0 + half * 8 + r;
        if (rr < M) {
            if (bcol0 < Nn) {
                float v = acc0[r];
                if (bias) v += bias[bcol0];
                if (act == 1) v = fmaxf(v, 0.f);
                C[(size_t)rr * Nn + bcol0] = v;
            }
            if (bcol1 < Nn) {
                float v = acc1[r];
                if (bias) v += bias[bcol1];
                if (act == 1) v = fmaxf(v, 0.f);
                C[(size_t)rr * Nn + bcol1] = v;
            }
        }
    }
}

// ---------------------------------------------------------------------------
// weight transpose+pad:  wt[col*Kp + k] = (k < K) ? (bf16)w[k*Nn + col] : 0
// ---------------------------------------------------------------------------
__global__ void k_cvt_t(const float* __restrict__ w, bf16* __restrict__ wt,
                        int K, int Nn, int Kp)
{
    int i = blockIdx.x * blockDim.x + threadIdx.x;
    if (i >= Nn * Kp) return;
    int col = i / Kp, k = i % Kp;
    wt[i] = (k < K) ? (bf16)w[(size_t)k * Nn + col] : (bf16)0.f;
}

// ---------------------------------------------------------------------------
// small utility kernels
// ---------------------------------------------------------------------------
__global__ void k_fill(float* __restrict__ p, float v, size_t n)
{
    size_t i = (size_t)blockIdx.x * blockDim.x + threadIdx.x;
    if (i < n) p[i] = v;
}

__global__ void k_add(float* __restrict__ a, const float* __restrict__ b, size_t n)
{
    size_t i = (size_t)blockIdx.x * blockDim.x + threadIdx.x;
    if (i < n) a[i] += b[i];
}

// tokens: tok[(nl*5+t)*4 + c] = vel[t,N0+nl,c];  tok[..3] = t/4
__global__ void k_build_tok(const float* __restrict__ vel, float* __restrict__ tok,
                            int n0, int NC, int N)
{
    int i = blockIdx.x * blockDim.x + threadIdx.x;
    if (i >= NC * 5) return;
    int nl = i / 5, t = i % 5, n = n0 + nl;
    float* o = tok + (size_t)i * 4;
    const float* v = vel + ((size_t)t * N + n) * 3;
    o[0] = v[0]; o[1] = v[1]; o[2] = v[2]; o[3] = t * 0.25f;
}

// per (node,head) 5-token attention; qkv row layout [q(64)|k(64)|v(64)]
__global__ void k_attn5(const float* __restrict__ qkv, float* __restrict__ o, int NC)
{
    int id = blockIdx.x * blockDim.x + threadIdx.x;
    if (id >= NC * 4) return;
    int n = id >> 2, h = id & 3;
    const float* base = qkv + (size_t)n * 5 * 192;
    const float scale = 0.25f;  // 1/sqrt(16)
    for (int i = 0; i < 5; ++i) {
        float q[16];
        for (int d = 0; d < 16; ++d) q[d] = base[i * 192 + h * 16 + d];
        float s[5], mx = -1e30f;
        for (int j = 0; j < 5; ++j) {
            float acc = 0.f;
            for (int d = 0; d < 16; ++d) acc += q[d] * base[j * 192 + 64 + h * 16 + d];
            s[j] = acc * scale; mx = fmaxf(mx, s[j]);
        }
        float den = 0.f;
        for (int j = 0; j < 5; ++j) { s[j] = __expf(s[j] - mx); den += s[j]; }
        float inv = 1.f / den;
        for (int d = 0; d < 16; ++d) {
            float acc = 0.f;
            for (int j = 0; j < 5; ++j) acc += s[j] * base[j * 192 + 128 + h * 16 + d];
            o[(size_t)(n * 5 + i) * 64 + h * 16 + d] = acc * inv;
        }
    }
}

// x = LayerNorm(x + r) * g + b   (post-norm residual), row width D
__global__ void k_add_ln(float* __restrict__ x, const float* __restrict__ r,
                         const float* __restrict__ g, const float* __restrict__ b,
                         int M, int D)
{
    int row = blockIdx.x * blockDim.x + threadIdx.x;
    if (row >= M) return;
    float* xp = x + (size_t)row * D;
    const float* rp = r + (size_t)row * D;
    float mu = 0.f;
    for (int d = 0; d < D; ++d) { float v = xp[d] + rp[d]; xp[d] = v; mu += v; }
    mu /= D;
    float var = 0.f;
    for (int d = 0; d < D; ++d) { float t = xp[d] - mu; var += t * t; }
    var /= D;
    float inv = rsqrtf(var + 1e-5f);
    for (int d = 0; d < D; ++d) xp[d] = g[d] * (xp[d] - mu) * inv + b[d];
}

__global__ void k_mean_T(const float* __restrict__ x, float* __restrict__ emb,
                         int n0, int NC)
{
    int i = blockIdx.x * blockDim.x + threadIdx.x;
    if (i >= NC * 64) return;
    int nl = i / 64, d = i % 64;
    float s = 0.f;
    for (int t = 0; t < 5; ++t) s += x[(size_t)(nl * 5 + t) * 64 + d];
    emb[(size_t)(n0 + nl) * 64 + d] = s * 0.2f;
}

__global__ void k_vort(const float* __restrict__ vel, float* __restrict__ vort, int N)
{
    int n = blockIdx.x * blockDim.x + threadIdx.x;
    if (n >= N) return;
    float s = 0.f;
    for (int c = 0; c < 3; ++c) {
        float d = vel[((size_t)4 * N + n) * 3 + c] - vel[(size_t)n * 3 + c];
        s += d * d;
    }
    vort[n] = sqrtf(s);
}

__global__ void k_nodefeat(const float* __restrict__ emb, const float* __restrict__ wall,
                           const float* __restrict__ air, float* __restrict__ nf, int N)
{
    int i = blockIdx.x * blockDim.x + threadIdx.x;
    if (i >= N * 66) return;
    int n = i / 66, f = i % 66;
    nf[i] = (f < 64) ? emb[(size_t)n * 64 + f] : ((f == 64) ? wall[n] : air[n]);
}

// ---- GAT pieces ----
__global__ void k_gat_coeff(const float* __restrict__ hlin, const float* __restrict__ asrc,
                            const float* __restrict__ adst, float* __restrict__ as_,
                            float* __restrict__ ad_, int N, int H, int dh)
{
    int id = blockIdx.x * blockDim.x + threadIdx.x;
    if (id >= N * H) return;
    int n = id / H, h = id % H;
    const float* hp = hlin + (size_t)n * H * dh + h * dh;
    float sa = 0.f, sd = 0.f;
    for (int d = 0; d < dh; ++d) { sa += hp[d] * asrc[h * dh + d]; sd += hp[d] * adst[h * dh + d]; }
    as_[id] = sa; ad_[id] = sd;
}

__device__ __forceinline__ void atomicMaxF(float* addr, float val)
{
    if (val >= 0.f) atomicMax((int*)addr, __float_as_int(val));
    else            atomicMin((unsigned int*)addr, __float_as_uint(val));
}

__global__ void k_edge_logit(const int* __restrict__ src, const int* __restrict__ dst,
                             const float* __restrict__ as_, const float* __restrict__ ad_,
                             float* __restrict__ elog, float* __restrict__ m, int E, int H)
{
    size_t id = (size_t)blockIdx.x * blockDim.x + threadIdx.x;
    if (id >= (size_t)E * H) return;
    int e = (int)(id / H), h = (int)(id % H);
    int s = src[e], d = dst[e];
    float l = as_[(size_t)s * H + h] + ad_[(size_t)d * H + h];
    l = (l > 0.f) ? l : 0.2f * l;            // leaky_relu(0.2)
    elog[id] = l;
    atomicMaxF(&m[(size_t)d * H + h], l);
}

__global__ void k_edge_exp(const int* __restrict__ dst, float* __restrict__ elog,
                           const float* __restrict__ m, float* __restrict__ den, int E, int H)
{
    size_t id = (size_t)blockIdx.x * blockDim.x + threadIdx.x;
    if (id >= (size_t)E * H) return;
    int e = (int)(id / H), h = (int)(id % H);
    int d = dst[e];
    float mv = m[(size_t)d * H + h];
    if (mv < -1e37f) mv = 0.f;               // where(isfinite(m), m, 0)
    float ex = __expf(elog[id] - mv);
    elog[id] = ex;
    atomicAdd(&den[(size_t)d * H + h], ex);
}

// one block per edge, blockDim = F = H*dh (coalesced over features)
__global__ void k_edge_aggr(const int* __restrict__ src, const int* __restrict__ dst,
                            const float* __restrict__ hlin, const float* __restrict__ elog,
                            const float* __restrict__ den, float* __restrict__ out,
                            int H, int dh)
{
    int e = blockIdx.x, f = threadIdx.x, F = blockDim.x;
    int h = f / dh;
    int s = src[e], d = dst[e];
    float alpha = elog[(size_t)e * H + h] / (den[(size_t)d * H + h] + 1e-16f);
    atomicAdd(&out[(size_t)d * F + f], hlin[(size_t)s * F + f] * alpha);
}

// out = elu( LN(out + bias) * g + b ), row width F
__global__ void k_gat_post(float* __restrict__ out, const float* __restrict__ bvec,
                           const float* __restrict__ g, const float* __restrict__ b,
                           int N, int F)
{
    int n = blockIdx.x * blockDim.x + threadIdx.x;
    if (n >= N) return;
    float* p = out + (size_t)n * F;
    float mu = 0.f;
    for (int f = 0; f < F; ++f) { float v = p[f] + bvec[f]; p[f] = v; mu += v; }
    mu /= F;
    float var = 0.f;
    for (int f = 0; f < F; ++f) { float t = p[f] - mu; var += t * t; }
    var /= F;
    float inv = rsqrtf(var + 1e-5f);
    for (int f = 0; f < F; ++f) {
        float v = g[f] * (p[f] - mu) * inv + b[f];
        p[f] = (v > 0.f) ? v : (__expf(v) - 1.f);
    }
}

__global__ void k_colmean(const float* __restrict__ h, float* __restrict__ gmean, int N, int F)
{
    int f = threadIdx.x;
    int n0 = blockIdx.x * 128;
    float s = 0.f;
    for (int i = 0; i < 128; ++i) { int n = n0 + i; if (n < N) s += h[(size_t)n * F + f]; }
    atomicAdd(&gmean[f], s);
}

__global__ void k_concat(const float* __restrict__ h, const float* __restrict__ gmean,
                         float* __restrict__ Hc, int N)
{
    size_t i = (size_t)blockIdx.x * blockDim.x + threadIdx.x;
    if (i >= (size_t)N * 256) return;
    int n = (int)(i / 256), f = (int)(i % 256);
    Hc[i] = (f < 128) ? h[(size_t)n * 128 + f] : gmean[f - 128] * (1.f / 50000.f);
}

__global__ void k_blend(const float* __restrict__ turb, const float* __restrict__ lam,
                        const float* __restrict__ wall, const float* __restrict__ vort,
                        float* __restrict__ out, int N)
{
    int n = blockIdx.x * blockDim.x + threadIdx.x;
    if (n >= N) return;
    float wd = wall[n] / (1.f + 1e-8f);
    float vo = vort[n] / (1.f + 1e-8f);
    float g = 1.f / (1.f + __expf(-((0.1f - wd) * 20.f + vo * 5.f)));
    for (int j = 0; j < 15; ++j) {
        float m = g * turb[(size_t)n * 15 + j] + (1.f - g) * lam[(size_t)n * 15 + j];
        int t = j / 3, c = j % 3;
        out[((size_t)t * N + n) * 3 + c] = m;   // (5, N, 3)
    }
}

// ---------------------------------------------------------------------------
// host orchestration
// ---------------------------------------------------------------------------
extern "C" void kernel_launch(void* const* d_in, const int* in_sizes, int n_in,
                              void* d_out, int out_size, void* d_ws, size_t ws_size,
                              hipStream_t stream)
{
    (void)in_sizes; (void)n_in; (void)out_size; (void)ws_size;
    const int N = 50000, E1 = 800000, E2 = 1600000;

    // pytree (sorted-key) flattening of params, then remaining inputs in dict order
    // gat: W, a_dst, a_src, b, ln_b, ln_g
    // enc: b1, b2, bo, bqkv, ln1b, ln1g, ln2b, ln2g, w1, w2, wo, wqkv
    const int I_BB1 = 0, I_BB2 = 6, I_E0 = 12, I_E1 = 24, I_LAM1 = 36, I_LAM2 = 42;
    const int I_PROJ_B = 48, I_PROJ_W = 49, I_TG = 50;          // tg1..4: 50,56,62,68
    const int I_TIN_B = 74, I_TIN_W = 75, I_TOUT_B = 76, I_TOUT_W = 77;
    const int I_VEL = 78, I_WALL = 79, I_AIR = 80, I_EI = 81, I_EID = 82;

    auto F32 = [&](int i) { return (const float*)d_in[i]; };
    const int* ei_s  = (const int*)d_in[I_EI];   const int* ei_d  = ei_s + E1;
    const int* eid_s = (const int*)d_in[I_EID];  const int* eid_d = eid_s + E2;
    const float* vel = F32(I_VEL);
    const float* wall = F32(I_WALL);
    const float* air = F32(I_AIR);
    float* out = (float*)d_out;

    // --- bf16 transposed+padded weight arena at front of ws (4 MB reserved) ---
    bf16* wb = (bf16*)d_ws;
    size_t wboff = 0;
    auto cvt = [&](int idx, int K, int Nn) -> const bf16* {
        int Kp = (K + 31) & ~31;
        bf16* dst = wb + wboff; wboff += (size_t)Nn * Kp;
        int n = Nn * Kp;
        k_cvt_t<<<(n + 255) / 256, 256, 0, stream>>>(F32(idx), dst, K, Nn, Kp);
        return dst;
    };
    const bf16* W_proj = cvt(I_PROJ_W, 4, 64);
    const bf16* W_qkv[2] = { cvt(I_E0 + 11, 64, 192), cvt(I_E1 + 11, 64, 192) };
    const bf16* W_o[2]   = { cvt(I_E0 + 10, 64, 64),  cvt(I_E1 + 10, 64, 64) };
    const bf16* W_1[2]   = { cvt(I_E0 + 8, 64, 128),  cvt(I_E1 + 8, 64, 128) };
    const bf16* W_2[2]   = { cvt(I_E0 + 9, 128, 64),  cvt(I_E1 + 9, 128, 64) };
    const bf16* W_bb1  = cvt(I_BB1, 66, 128);
    const bf16* W_bb2  = cvt(I_BB2, 128, 128);
    const bf16* W_lam1 = cvt(I_LAM1, 256, 64);
    const bf16* W_lam2 = cvt(I_LAM2, 64, 15);
    const bf16* W_tin  = cvt(I_TIN_W, 256, 256);
    const bf16* W_tg[4] = { cvt(I_TG, 256, 256), cvt(I_TG + 6, 256, 256),
                            cvt(I_TG + 12, 256, 256), cvt(I_TG + 18, 256, 256) };
    const bf16* W_tout = cvt(I_TOUT_W, 256, 15);

    // --- fp32 scratch arena (~276 MB) ---
    float* fb = (float*)((char*)d_ws + (4 << 20));
    size_t off = 0;
    auto alloc = [&](size_t n) { float* p = fb + off; off += n; return p; };
    float* vort  = alloc(N);
    float* gmean = alloc(256);
    float* s_as  = alloc((size_t)N * 4);
    float* s_ad  = alloc((size_t)N * 4);
    float* s_m   = alloc((size_t)N * 4);
    float* s_den = alloc((size_t)N * 4);
    float* elog  = alloc((size_t)E2 * 4);
    float* temb  = alloc((size_t)N * 64);
    float* nf    = alloc((size_t)N * 66);
    float* lamo  = alloc((size_t)N * 64);
    float* lam2o = alloc((size_t)N * 16);
    float* s0 = alloc((size_t)N * 256);   // H_cat      (encoder: tok + ffn scratch)
    float* s1 = alloc((size_t)N * 256);   // xt         (encoder: x)
    float* s2 = alloc((size_t)N * 256);   // hlin       (encoder: qkv)
    float* s3 = alloc((size_t)N * 256);   // hout       (encoder: attn)
    (void)alloc(256);                     // guard pad: GEMM A-tail overread

    auto gemm = [&](const float* A, const bf16* Wt, const float* bias, float* C,
                    int M, int K, int Nn, int act) {
        int tiles = ((M + 15) / 16) * ((Nn + 31) / 32);
        k_gemm_wmma<<<(tiles + 7) / 8, 256, 0, stream>>>(A, Wt, bias, C, M, K, Nn, act);
    };

    // ---------------- temporal encoder (chunks of 10k nodes) ----------------
    const int NC = 10000, MC = NC * 5;
    float* tok = s0;
    float* f1b = s0 + 1000000;
    float* xb = s1, *qb = s2, *ab = s3;
    for (int c = 0; c < 5; ++c) {
        int n0 = c * NC;
        k_build_tok<<<(NC * 5 + 255) / 256, 256, 0, stream>>>(vel, tok, n0, NC, N);
        gemm(tok, W_proj, F32(I_PROJ_B), xb, MC, 4, 64, 0);
        for (int l = 0; l < 2; ++l) {
            int base = l ? I_E1 : I_E0;
            gemm(xb, W_qkv[l], F32(base + 3), qb, MC, 64, 192, 0);
            k_attn5<<<(NC * 4 + 255) / 256, 256, 0, stream>>>(qb, ab, NC);
            gemm(ab, W_o[l], F32(base + 2), qb, MC, 64, 64, 0);
            k_add_ln<<<(MC + 255) / 256, 256, 0, stream>>>(xb, qb, F32(base + 5), F32(base + 4), MC, 64);
            gemm(xb, W_1[l], F32(base + 0), f1b, MC, 64, 128, 1);
            gemm(f1b, W_2[l], F32(base + 1), qb, MC, 128, 64, 0);
            k_add_ln<<<(MC + 255) / 256, 256, 0, stream>>>(xb, qb, F32(base + 7), F32(base + 6), MC, 64);
        }
        k_mean_T<<<(NC * 64 + 255) / 256, 256, 0, stream>>>(xb, temb, n0, NC);
    }

    // ---------------- physics features ----------------
    k_vort<<<(N + 255) / 256, 256, 0, stream>>>(vel, vort, N);
    k_nodefeat<<<(N * 66 + 255) / 256, 256, 0, stream>>>(temb, wall, air, nf, N);

    // ---------------- GAT layer helper ----------------
    auto gat = [&](const float* xin, int cin, const bf16* Wt, int Hh, int dh,
                   int pb, const int* esrc, const int* edst, int E,
                   float* hlin, float* outb) {
        int F = Hh * dh;
        gemm(xin, Wt, nullptr, hlin, N, cin, F, 0);
        k_gat_coeff<<<(N * Hh + 255) / 256, 256, 0, stream>>>(
            hlin, F32(pb + 2), F32(pb + 1), s_as, s_ad, N, Hh, dh);
        k_fill<<<(N * Hh + 255) / 256, 256, 0, stream>>>(s_m, -3.3e38f, (size_t)N * Hh);
        k_fill<<<(N * Hh + 255) / 256, 256, 0, stream>>>(s_den, 0.f, (size_t)N * Hh);
        k_fill<<<(int)(((size_t)N * F + 255) / 256), 256, 0, stream>>>(outb, 0.f, (size_t)N * F);
        k_edge_logit<<<(int)(((size_t)E * Hh + 255) / 256), 256, 0, stream>>>(
            esrc, edst, s_as, s_ad, elog, s_m, E, Hh);
        k_edge_exp<<<(int)(((size_t)E * Hh + 255) / 256), 256, 0, stream>>>(
            edst, elog, s_m, s_den, E, Hh);
        k_edge_aggr<<<E, F, 0, stream>>>(esrc, edst, hlin, elog, s_den, outb, Hh, dh);
        k_gat_post<<<(N + 255) / 256, 256, 0, stream>>>(outb, F32(pb + 3), F32(pb + 5), F32(pb + 4), N, F);
    };

    // ---------------- shared backbone ----------------
    gat(nf, 66, W_bb1, 4, 32, I_BB1, ei_s, ei_d, E1, s2, s3);     // h1 = s3 (N,128)
    gat(s3, 128, W_bb2, 4, 32, I_BB2, ei_s, ei_d, E1, s2, s3);    // h2 = s3 (N,128)
    k_fill<<<1, 128, 0, stream>>>(gmean, 0.f, 128);
    k_colmean<<<(N + 127) / 128, 128, 0, stream>>>(s3, gmean, N, 128);
    k_concat<<<(int)(((size_t)N * 256 + 255) / 256), 256, 0, stream>>>(s3, gmean, s0, N); // H = s0

    // ---------------- laminar expert ----------------
    gat(s0, 256, W_lam1, 4, 16, I_LAM1, ei_s, ei_d, E1, s2, lamo);    // (N,64)
    gat(lamo, 64, W_lam2, 1, 15, I_LAM2, ei_s, ei_d, E1, s2, lam2o);  // (N,15)

    // ---------------- turbulent expert (dense graph) ----------------
    gemm(s0, W_tin, F32(I_TIN_B), s1, N, 256, 256, 0);                // xt = s1
    gat(s1, 256, W_tg[0], 4, 64, I_TG + 0,  eid_s, eid_d, E2, s2, s3);
    gat(s3, 256, W_tg[1], 4, 64, I_TG + 6,  eid_s, eid_d, E2, s2, s3);
    k_add<<<(int)(((size_t)N * 256 + 255) / 256), 256, 0, stream>>>(s1, s3, (size_t)N * 256);
    gat(s1, 256, W_tg[2], 4, 64, I_TG + 12, eid_s, eid_d, E2, s2, s3);
    gat(s3, 256, W_tg[3], 4, 64, I_TG + 18, eid_s, eid_d, E2, s2, s3);
    k_add<<<(int)(((size_t)N * 256 + 255) / 256), 256, 0, stream>>>(s1, s3, (size_t)N * 256);
    gemm(s1, W_tout, F32(I_TOUT_B), s2, N, 256, 15, 0);               // turb = s2 (N,15)

    // ---------------- gate + blend + transpose ----------------
    k_blend<<<(N + 255) / 256, 256, 0, stream>>>(s2, lam2o, wall, vort, out, N);
}
